// MultiHeadAttention_47528108097550
// MI455X (gfx1250) — matrix operand order repaired and checked by reading, not compiled, not run on hardware
//
#include <hip/hip_runtime.h>

// ---------------------------------------------------------------------------
// MHA forward for MI455X (gfx1250, wave32, WMMA 16x16x32 bf16, f32 accum).
// B=4 S=2048 E=768 H=12 D=64.  ~90 GFLOP; bf16 WMMA path, working sets in L2.
// ---------------------------------------------------------------------------

typedef __attribute__((ext_vector_type(16))) __bf16 v16bf;
typedef __attribute__((ext_vector_type(8)))  __bf16 v8bf;
typedef __attribute__((ext_vector_type(8)))  float  v8f;

#define E_  768
#define H_  12
#define D_  64
#define B_  4
#define S_  2048
#define E3_ (3 * E_)
#define NROW_ (B_ * S_)   // 8192 token rows

union Frag16 { v16bf v; v8bf h[2]; };

// WMMA wrapper: D = A(16x32 bf16) * B(32x16 bf16) + C(16x16 f32)
__device__ __forceinline__ v8f wmma_bf16(v16bf a, v16bf b, v8f c) {
    return __builtin_amdgcn_wmma_f32_16x16x32_bf16(
        /*neg_a=*/false, a, /*neg_b=*/false, b,
        /*c_mod=*/(short)0, c, /*reuse_a=*/false, /*reuse_b=*/false);
}

// A-matrix fragment (16 rows x 32 K) from row-major [., lda] at (rowBase,kBase).
// ISA layout: lane m=L&15, hi=L>>4; VGPR v: k = (v>>2)*16 + hi*8 + (v&3)*2.
// => two contiguous 16B loads per lane at k = kBase+hi*8 and kBase+16+hi*8.
__device__ __forceinline__ v16bf load_a_frag(const __bf16* base, int lda,
                                             int rowBase, int kBase, int lane) {
    const int m = lane & 15, hi = lane >> 4;
    const __bf16* p = base + (size_t)(rowBase + m) * lda + kBase;
    Frag16 f;
    f.h[0] = *(const v8bf*)(p + hi * 8);
    f.h[1] = *(const v8bf*)(p + 16 + hi * 8);
    return f.v;
}

// B-matrix fragment (32 K x 16 N) where element (k,n) = mem[(colBase+n)*ldb + kBase+k].
// ISA layout: lane n=L&15, hi=L>>4; same k order as A: k = hi*8 + s (s<8), 16 + hi*8 + (s-8) (s>=8).
__device__ __forceinline__ v16bf load_b_frag(const __bf16* base, int ldb,
                                             int colBase, int kBase, int lane) {
    const int n = lane & 15, hi = lane >> 4;
    const __bf16* p = base + (size_t)(colBase + n) * ldb + kBase;
    Frag16 f;
    f.h[0] = *(const v8bf*)(p + hi * 8);
    f.h[1] = *(const v8bf*)(p + 16 + hi * 8);
    return f.v;
}

// ---------------------------------------------------------------------------
__global__ void mha_cvt_bf16(const float* __restrict__ in,
                             __bf16* __restrict__ out, int n) {
    int i = blockIdx.x * blockDim.x + threadIdx.x;
    int stride = gridDim.x * blockDim.x;
    for (; i < n; i += stride) out[i] = (__bf16)in[i];
}

// ---------------------------------------------------------------------------
// QKV projection: qkv[row, f] = sum_e x[row,e] * W[f,e] + bias[f]
// One wave -> 16x64 output tile. Scatter into Q,K:[B,H,S,D]  V^T:[B,H,D,S].
// Q gets the 1/sqrt(D) softmax scale folded in.
__global__ void mha_qkv_gemm(const __bf16* __restrict__ xb,
                             const __bf16* __restrict__ w,
                             const float* __restrict__ bias,
                             __bf16* __restrict__ Q,
                             __bf16* __restrict__ K,
                             __bf16* __restrict__ Vt) {
    const int lane = threadIdx.x & 31;
    const int wave = threadIdx.x >> 5;
    const int nT64 = E3_ / 64;                      // 36 column tiles
    const int flat = blockIdx.x * 8 + wave;
    const int mBase = (flat / nT64) * 16;
    const int nBase = (flat % nT64) * 64;

    v8f acc[4] = {};
    for (int k = 0; k < E_; k += 32) {
        v16bf a = load_a_frag(xb, E_, mBase, k, lane);
#pragma unroll
        for (int t = 0; t < 4; t++) {
            v16bf b = load_b_frag(w, E_, nBase + t * 16, k, lane);
            acc[t] = wmma_bf16(a, b, acc[t]);
        }
    }

    const int n = lane & 15, hi = lane >> 4;
    const float qScale = 0.125f;                    // 1/sqrt(64)
#pragma unroll
    for (int t = 0; t < 4; t++) {
        const int col = nBase + t * 16 + n;
        const float bv = bias[col];
#pragma unroll
        for (int r = 0; r < 8; r++) {
            const int row = mBase + hi * 8 + r;
            const int bb = row / S_, ss = row % S_;
            float v = acc[t][r] + bv;
            if (col < E_) {
                const int h = col / D_, d = col % D_;
                Q[((size_t)(bb * H_ + h) * S_ + ss) * D_ + d] = (__bf16)(v * qScale);
            } else if (col < 2 * E_) {
                const int e = col - E_, h = e / D_, d = e % D_;
                K[((size_t)(bb * H_ + h) * S_ + ss) * D_ + d] = (__bf16)v;
            } else {
                const int e = col - 2 * E_, h = e / D_, d = e % D_;
                Vt[((size_t)(bb * H_ + h) * D_ + d) * S_ + ss] = (__bf16)v;
            }
        }
    }
}

// ---------------------------------------------------------------------------
// Flash attention: one wave owns a 16-row q tile; kv tiles of 32.
// scores = Q*K^T (already scaled), +1.0 on allowed / -BIG on masked (faithful),
// online softmax, ctx += P * V via LDS re-layout of P (C-format -> A-format).
// ctx written as [B, S, H, D]  (== [B,S,E] for the out-proj GEMM).
__global__ void mha_attn(const __bf16* __restrict__ Q,
                         const __bf16* __restrict__ K,
                         const __bf16* __restrict__ Vt,
                         __bf16* __restrict__ ctx) {
    __shared__ __align__(16) __bf16 pbuf[4][16 * 32];   // per-wave P staging

    const int lane = threadIdx.x & 31;
    const int wave = threadIdx.x >> 5;
    const int bh = blockIdx.y;                      // b*H + h
    const int bb = bh / H_, hh = bh % H_;
    const int qBase = (blockIdx.x * 4 + wave) * 16;
    const int nlo = lane & 15, hi = lane >> 4;

    const __bf16* Qh = Q  + (size_t)bh * S_ * D_;
    const __bf16* Kh = K  + (size_t)bh * S_ * D_;
    const __bf16* Vh = Vt + (size_t)bh * D_ * S_;
    __bf16* pw = pbuf[wave];

    // Q fragments for d-chunks [0,32) and [32,64)
    const v16bf qa0 = load_a_frag(Qh, D_, qBase, 0, lane);
    const v16bf qa1 = load_a_frag(Qh, D_, qBase, 32, lane);

    v8f o[4] = {};
    float mrow[8], lrow[8];
#pragma unroll
    for (int r = 0; r < 8; r++) { mrow[r] = -3.0e38f; lrow[r] = 0.0f; }

    for (int jBase = 0; jBase <= qBase + 15; jBase += 32) {
        // ---- scores: 16(q) x 32(kv), two 16-col C tiles -------------------
        v8f s0 = {}, s1 = {};
        {
            v16bf b0 = load_b_frag(Kh, D_, jBase,      0, lane);
            v16bf b1 = load_b_frag(Kh, D_, jBase + 16, 0, lane);
            s0 = wmma_bf16(qa0, b0, s0);
            s1 = wmma_bf16(qa0, b1, s1);
            b0 = load_b_frag(Kh, D_, jBase,      32, lane);
            b1 = load_b_frag(Kh, D_, jBase + 16, 32, lane);
            s0 = wmma_bf16(qa1, b0, s0);
            s1 = wmma_bf16(qa1, b1, s1);
        }
        // ---- causal mask (faithful: +1.0 allowed, -inf masked) ------------
#pragma unroll
        for (int r = 0; r < 8; r++) {
            const int qrow = qBase + hi * 8 + r;
            s0[r] = (jBase + nlo      <= qrow) ? (s0[r] + 1.0f) : -3.0e38f;
            s1[r] = (jBase + 16 + nlo <= qrow) ? (s1[r] + 1.0f) : -3.0e38f;
        }
        // ---- online softmax: rows live across 16-lane groups --------------
        float p0[8], p1[8], csc[8];
#pragma unroll
        for (int r = 0; r < 8; r++) {
            float t = fmaxf(s0[r], s1[r]);
            t = fmaxf(t, __shfl_xor(t, 1, 32));
            t = fmaxf(t, __shfl_xor(t, 2, 32));
            t = fmaxf(t, __shfl_xor(t, 4, 32));
            t = fmaxf(t, __shfl_xor(t, 8, 32));
            const float mn = fmaxf(mrow[r], t);
            const float sc = __expf(mrow[r] - mn);
            mrow[r] = mn;
            const float e0 = __expf(s0[r] - mn);
            const float e1 = __expf(s1[r] - mn);
            float rs = e0 + e1;
            rs += __shfl_xor(rs, 1, 32);
            rs += __shfl_xor(rs, 2, 32);
            rs += __shfl_xor(rs, 4, 32);
            rs += __shfl_xor(rs, 8, 32);
            lrow[r] = lrow[r] * sc + rs;
            csc[r] = sc;  p0[r] = e0;  p1[r] = e1;
        }
        // rescale running ctx accumulators (same row mapping as C layout)
#pragma unroll
        for (int t = 0; t < 4; t++)
#pragma unroll
            for (int r = 0; r < 8; r++) o[t][r] *= csc[r];

        // ---- P: C-layout -> LDS -> A-layout fragment ----------------------
#pragma unroll
        for (int r = 0; r < 8; r++) {
            pw[(hi * 8 + r) * 32 + nlo]      = (__bf16)p0[r];
            pw[(hi * 8 + r) * 32 + 16 + nlo] = (__bf16)p1[r];
        }
        const v16bf pa = load_a_frag(pw, 32, 0, 0, lane);   // wave-private, DS in-order

        // ---- ctx += P(16x32) * V(32x64): V^T rows are contiguous in kv ----
#pragma unroll
        for (int t = 0; t < 4; t++) {
            v16bf bv = load_b_frag(Vh, S_, t * 16, jBase, lane);
            o[t] = wmma_bf16(pa, bv, o[t]);
        }
    }

    // normalize, store ctx as [B, S, H, D]
#pragma unroll
    for (int r = 0; r < 8; r++) lrow[r] = 1.0f / lrow[r];
#pragma unroll
    for (int t = 0; t < 4; t++) {
#pragma unroll
        for (int r = 0; r < 8; r++) {
            const int q = qBase + hi * 8 + r;
            const int d = t * 16 + nlo;
            ctx[(((size_t)bb * S_ + q) * H_ + hh) * D_ + d] = (__bf16)(o[t][r] * lrow[r]);
        }
    }
}

// ---------------------------------------------------------------------------
// Output projection: out[row,f] = sum_e ctx[row,e]*Wout[f,e] + bias[f]  (f32 out)
__global__ void mha_out_gemm(const __bf16* __restrict__ ctxb,
                             const __bf16* __restrict__ w,
                             const float* __restrict__ bias,
                             float* __restrict__ out) {
    const int lane = threadIdx.x & 31;
    const int wave = threadIdx.x >> 5;
    const int nT64 = E_ / 64;                       // 12 column tiles
    const int flat = blockIdx.x * 8 + wave;
    const int mBase = (flat / nT64) * 16;
    const int nBase = (flat % nT64) * 64;

    v8f acc[4] = {};
    for (int k = 0; k < E_; k += 32) {
        v16bf a = load_a_frag(ctxb, E_, mBase, k, lane);
#pragma unroll
        for (int t = 0; t < 4; t++) {
            v16bf b = load_b_frag(w, E_, nBase + t * 16, k, lane);
            acc[t] = wmma_bf16(a, b, acc[t]);
        }
    }
    const int n = lane & 15, hi = lane >> 4;
#pragma unroll
    for (int t = 0; t < 4; t++) {
        const int col = nBase + t * 16 + n;
        const float bv = bias[col];
#pragma unroll
        for (int r = 0; r < 8; r++) {
            const int row = mBase + hi * 8 + r;
            out[(size_t)row * E_ + col] = acc[t][r] + bv;
        }
    }
}

// ---------------------------------------------------------------------------
extern "C" void kernel_launch(void* const* d_in, const int* in_sizes, int n_in,
                              void* d_out, int out_size, void* d_ws, size_t ws_size,
                              hipStream_t stream) {
    const float* x         = (const float*)d_in[0];   // [B,S,E]
    const float* in_proj_w = (const float*)d_in[1];   // [3E,E]
    const float* in_proj_b = (const float*)d_in[2];   // [3E]
    const float* out_w     = (const float*)d_in[3];   // [E,E]
    const float* out_b     = (const float*)d_in[4];   // [E]
    float* out = (float*)d_out;                       // [B,S,E] f32

    // workspace layout (256B-aligned chunks), total ~69 MB
    char* ws = (char*)d_ws;
    size_t off = 0;
    auto take = [&](size_t bytes) -> void* {
        void* p = ws + off;
        off += (bytes + 255) & ~(size_t)255;
        return p;
    };
    __bf16* xb   = (__bf16*)take((size_t)NROW_ * E_ * 2);      // x bf16
    __bf16* wqkv = (__bf16*)take((size_t)E3_ * E_ * 2);        // in_proj_w bf16
    __bf16* wout = (__bf16*)take((size_t)E_ * E_ * 2);         // out_w bf16
    __bf16* Qb   = (__bf16*)take((size_t)B_ * H_ * S_ * D_ * 2);
    __bf16* Kb   = (__bf16*)take((size_t)B_ * H_ * S_ * D_ * 2);
    __bf16* Vtb  = (__bf16*)take((size_t)B_ * H_ * D_ * S_ * 2);
    __bf16* ctxb = (__bf16*)take((size_t)NROW_ * E_ * 2);
    (void)ws_size; (void)in_sizes; (void)n_in; (void)out_size;

    // 1) convert inputs to bf16
    {
        int n = NROW_ * E_;
        mha_cvt_bf16<<<(n + 255) / 256, 256, 0, stream>>>(x, xb, n);
        n = E3_ * E_;
        mha_cvt_bf16<<<(n + 255) / 256, 256, 0, stream>>>(in_proj_w, wqkv, n);
        n = E_ * E_;
        mha_cvt_bf16<<<(n + 255) / 256, 256, 0, stream>>>(out_w, wout, n);
    }
    // 2) QKV projection: (8192/16)*(2304/64) = 18432 waves / 8 per block
    mha_qkv_gemm<<<18432 / 8, 256, 0, stream>>>(xb, wqkv, in_proj_b, Qb, Kb, Vtb);
    // 3) attention: grid = (S/16/4, B*H), 4 waves/block
    mha_attn<<<dim3(S_ / 64, B_ * H_), 128, 0, stream>>>(Qb, Kb, Vtb, ctxb);
    // 4) output projection: (8192/16)*(768/64) = 6144 waves / 8 per block
    mha_out_gemm<<<6144 / 8, 256, 0, stream>>>(ctxb, wout, out_b, out);
}
